// EncSALayer_21878563406038
// MI455X (gfx1250) — matrix operand order repaired and checked by reading, not compiled, not run on hardware
//
#include <hip/hip_runtime.h>
#include <hip/hip_bf16.h>
#include <cstdint>

#define B_  4
#define S_  2048
#define C_  1024
#define H_  16
#define D_  64
#define KW_ 9
#define F4_ 4096
#define BS_ (B_ * S_)

typedef __bf16 bf16_t;
typedef __attribute__((ext_vector_type(16))) __bf16 v16bf;
typedef __attribute__((ext_vector_type(8)))  float  v8f;

// ---------------------------------------------------------------------------
// WMMA helpers (CDNA5, wave32): D = A(16x32 bf16) x B(32x16 bf16) + C(16x16 f32)
// A layout: lane L -> row m=L&15; K elems kb..kb+7 and kb+16..kb+23, kb=(L>>4)*8
// B layout: lane L -> col n=L&15; K elems kb..kb+7 and kb+16..kb+23, kb=(L>>4)*8
// C/D layout: vgpr r, lane L -> (m = r + 8*(L>>4), n = L&15)
// ---------------------------------------------------------------------------
__device__ __forceinline__ v8f acc_zero() {
  v8f a;
#pragma unroll
  for (int i = 0; i < 8; ++i) a[i] = 0.0f;
  return a;
}

__device__ __forceinline__ v16bf load_fragA(const bf16_t* p) {
  union { v16bf v; float2 f[4]; } u;
  u.f[0] = *reinterpret_cast<const float2*>(p);
  u.f[1] = *reinterpret_cast<const float2*>(p + 4);
  u.f[2] = *reinterpret_cast<const float2*>(p + 16);
  u.f[3] = *reinterpret_cast<const float2*>(p + 20);
  return u.v;
}

__device__ __forceinline__ v16bf load_fragB(const bf16_t* p) {
  union { v16bf v; float2 f[4]; } u;
  u.f[0] = *reinterpret_cast<const float2*>(p);
  u.f[1] = *reinterpret_cast<const float2*>(p + 4);
  u.f[2] = *reinterpret_cast<const float2*>(p + 16);
  u.f[3] = *reinterpret_cast<const float2*>(p + 20);
  return u.v;
}

__device__ __forceinline__ v8f wmma_bf16(v16bf a, v16bf b, v8f c) {
  return __builtin_amdgcn_wmma_f32_16x16x32_bf16(false, a, false, b, (short)0, c,
                                                 false, false);
}

// ---------------------------------------------------------------------------
// Weight conversion / repack
// ---------------------------------------------------------------------------
__global__ void f32_to_bf16_kernel(const float* __restrict__ src,
                                   bf16_t* __restrict__ dst, size_t n) {
  size_t i = (size_t)blockIdx.x * 256 + threadIdx.x;
  if (i < n) dst[i] = (bf16_t)src[i];
}

// ffn1_w [F][C][K] -> bf16 [K][F][C]  (each tap becomes a row-major [F,C] weight)
__global__ void repack_ffn1_kernel(const float* __restrict__ src,
                                   bf16_t* __restrict__ dst) {
  size_t i = (size_t)blockIdx.x * 256 + threadIdx.x;
  const size_t total = (size_t)F4_ * C_ * KW_;
  if (i >= total) return;
  int k = (int)(i % KW_);
  size_t t = i / KW_;
  int c = (int)(t % C_);
  int f = (int)(t / C_);
  dst[((size_t)k * F4_ + f) * C_ + c] = (bf16_t)src[i];
}

// ---------------------------------------------------------------------------
// LayerNorm: one block per row (C=1024), fp32 in -> bf16 out
// ---------------------------------------------------------------------------
__global__ __launch_bounds__(256) void layernorm_kernel(
    const float* __restrict__ X, const float* __restrict__ w,
    const float* __restrict__ b, bf16_t* __restrict__ Y) {
  __shared__ float red0[8];
  __shared__ float red1[8];
  const int row = blockIdx.x;
  const float* x = X + (size_t)row * C_;
  float s = 0.0f, s2 = 0.0f;
  for (int c = threadIdx.x; c < C_; c += 256) {
    float v = x[c];
    s += v;
    s2 += v * v;
  }
#pragma unroll
  for (int off = 1; off < 32; off <<= 1) {
    s  += __shfl_xor(s,  off, 32);
    s2 += __shfl_xor(s2, off, 32);
  }
  const int wid = threadIdx.x >> 5, lane = threadIdx.x & 31;
  if (lane == 0) { red0[wid] = s; red1[wid] = s2; }
  __syncthreads();
  s = 0.0f; s2 = 0.0f;
#pragma unroll
  for (int i = 0; i < 8; ++i) { s += red0[i]; s2 += red1[i]; }
  const float mu  = s / (float)C_;
  const float var = s2 / (float)C_ - mu * mu;
  const float rstd = rsqrtf(var + 1e-5f);
  for (int c = threadIdx.x; c < C_; c += 256)
    Y[(size_t)row * C_ + c] = (bf16_t)((x[c] - mu) * rstd * w[c] + b[c]);
}

// ---------------------------------------------------------------------------
// Generic LDS-tiled bf16 GEMM: out[M,N] = A[M,K] * W[N,K]^T (+ epilogue)
// Block tile 128x128, BK=32, 8 waves as 4(m) x 2(n), 32x64 per wave (2x4 WMMA)
// Next-K-slice prefetch via global_prefetch_b8.
// EPI 0: plain fp32 store (qkv)
// EPI 1: (acc + resid) * nonpad          (out_proj)
// EPI 2: (acc + bias + resid) * nonpad   (ffn2, final output)
// ---------------------------------------------------------------------------
template <int EPI>
__global__ __launch_bounds__(256) void gemm_bf16_kernel(
    const bf16_t* __restrict__ A, const bf16_t* __restrict__ W,
    float* __restrict__ out, const float* __restrict__ resid,
    const float* __restrict__ bias, const int* __restrict__ padmask,
    int M, int N, int Kd) {
  __shared__ bf16_t sA[128][40];
  __shared__ bf16_t sB[128][40];
  const int tid = threadIdx.x;
  const int wid = tid >> 5, lane = tid & 31;
  const int wm = wid >> 1, wn = wid & 1;
  const int m0 = blockIdx.y * 128, n0 = blockIdx.x * 128;
  const int r = lane & 15;
  const int kbA = (lane >> 4) * 8, kbB = (lane >> 4) * 8;

  v8f acc[2][4];
#pragma unroll
  for (int i = 0; i < 2; ++i)
#pragma unroll
    for (int j = 0; j < 4; ++j) acc[i][j] = acc_zero();

  const int pr = tid >> 1, pc = (tid & 1) * 16;  // prefetch row/col
  for (int k0 = 0; k0 < Kd; k0 += 32) {
#pragma unroll
    for (int i = 0; i < 2; ++i) {
      int cid = tid + 256 * i;
      int rr = cid >> 2, cc = (cid & 3) << 3;
      *reinterpret_cast<float4*>(&sA[rr][cc]) =
          *reinterpret_cast<const float4*>(A + (size_t)(m0 + rr) * Kd + k0 + cc);
      *reinterpret_cast<float4*>(&sB[rr][cc]) =
          *reinterpret_cast<const float4*>(W + (size_t)(n0 + rr) * Kd + k0 + cc);
    }
    if (k0 + 32 < Kd) {  // pull next K-slice toward L0/L2 while we compute
      __builtin_prefetch(A + (size_t)(m0 + pr) * Kd + k0 + 32 + pc, 0, 1);
      __builtin_prefetch(W + (size_t)(n0 + pr) * Kd + k0 + 32 + pc, 0, 1);
    }
    __syncthreads();
    v16bf af[2], bfm[4];
#pragma unroll
    for (int i = 0; i < 2; ++i) af[i] = load_fragA(&sA[wm * 32 + i * 16 + r][kbA]);
#pragma unroll
    for (int j = 0; j < 4; ++j) bfm[j] = load_fragB(&sB[wn * 64 + j * 16 + r][kbB]);
#pragma unroll
    for (int i = 0; i < 2; ++i)
#pragma unroll
      for (int j = 0; j < 4; ++j) acc[i][j] = wmma_bf16(af[i], bfm[j], acc[i][j]);
    __syncthreads();
  }

  const int nIdx = lane & 15, mBase = (lane >> 4) * 8;
#pragma unroll
  for (int i = 0; i < 2; ++i)
#pragma unroll
    for (int j = 0; j < 4; ++j) {
      int nn = n0 + wn * 64 + j * 16 + nIdx;
#pragma unroll
      for (int e = 0; e < 8; ++e) {
        int mm = m0 + wm * 32 + i * 16 + mBase + e;
        size_t idx = (size_t)mm * N + nn;
        float v = acc[i][j][e];
        if (EPI == 0) {
          out[idx] = v;
        } else if (EPI == 1) {
          float np = 1.0f - (float)padmask[mm];
          out[idx] = (v + resid[idx]) * np;
        } else {
          float np = 1.0f - (float)padmask[mm];
          out[idx] = (v + bias[nn] + resid[idx]) * np;
        }
      }
    }
}

// ---------------------------------------------------------------------------
// RoPE + split qkv[B,S,3C] -> Q,K bf16 [B,H,S,D] (roped), V^T bf16 [B,H,D,S]
// ---------------------------------------------------------------------------
__global__ void rope_split_kernel(const float* __restrict__ qkv,
                                  bf16_t* __restrict__ Q,
                                  bf16_t* __restrict__ Kq,
                                  bf16_t* __restrict__ Vt) {
  size_t idx = (size_t)blockIdx.x * 256 + threadIdx.x;  // over B*S*H*(D/2)
  const size_t total = (size_t)B_ * S_ * H_ * (D_ / 2);
  if (idx >= total) return;
  int dp = (int)(idx & 31);           // D/2 = 32
  size_t t1 = idx >> 5;
  int h = (int)(t1 & 15);             // H = 16
  size_t t2 = t1 >> 4;
  int s = (int)(t2 & 2047);           // S = 2048
  int b = (int)(t2 >> 11);
  int d = dp * 2;

  size_t base = ((size_t)b * S_ + s) * (3 * C_) + h * D_ + d;
  float q0 = qkv[base],          q1 = qkv[base + 1];
  float k0 = qkv[base + C_],     k1 = qkv[base + C_ + 1];
  float v0 = qkv[base + 2 * C_], v1 = qkv[base + 2 * C_ + 1];

  float inv = powf(10000.0f, -(float)d / (float)D_);
  float ang = (float)s * inv;
  float co = cosf(ang), si = sinf(ang);

  size_t qi = (((size_t)b * H_ + h) * S_ + s) * D_ + d;
  Q[qi]     = (bf16_t)(q0 * co - q1 * si);
  Q[qi + 1] = (bf16_t)(q1 * co + q0 * si);
  Kq[qi]     = (bf16_t)(k0 * co - k1 * si);
  Kq[qi + 1] = (bf16_t)(k1 * co + k0 * si);

  size_t vi = (((size_t)b * H_ + h) * D_ + d) * S_ + s;
  Vt[vi]      = (bf16_t)v0;
  Vt[vi + S_] = (bf16_t)v1;
}

// ---------------------------------------------------------------------------
// Attention: one workgroup per (b, h, q-tile of 16). Full 16x2048 fp32 score
// block staged in LDS (128KB; CDNA5 WGP has 320KB), softmax in LDS, then
// P(16xS) x V(SxD) with WMMA. Deterministic (no atomics).
// ---------------------------------------------------------------------------
__global__ __launch_bounds__(256) void attn_kernel(
    const bf16_t* __restrict__ Q, const bf16_t* __restrict__ Km,
    const bf16_t* __restrict__ Vt, const int* __restrict__ padmask,
    bf16_t* __restrict__ ctx) {
  __shared__ float  sS[16 * S_];   // 128 KB scores (reused for ctx partials)
  __shared__ bf16_t sP[16 * S_];   //  64 KB probabilities
  __shared__ float  sRow[16];      // row sums

  const int tid = threadIdx.x, wid = tid >> 5, lane = tid & 31;
  const int qt = blockIdx.x, hh = blockIdx.y, bb = blockIdx.z;
  const int q0 = qt * 16;
  const size_t bh = (size_t)bb * H_ + hh;
  const int r = lane & 15;
  const int kbA = (lane >> 4) * 8, kbB = (lane >> 4) * 8;
  const int nIdx = lane & 15, mBase = (lane >> 4) * 8;

  // Q fragments: 16 rows x 64 (two K-chunks of 32)
  const bf16_t* Qbase = Q + (bh * S_ + q0) * D_;
  v16bf qf[2];
#pragma unroll
  for (int kc = 0; kc < 2; ++kc)
    qf[kc] = load_fragA(Qbase + (size_t)r * D_ + kc * 32 + kbA);

  // ---- scores: Q x K^T, scaled + key-padding mask, into LDS ----
  for (int kt = wid; kt < S_ / 16; kt += 8) {
    const bf16_t* Kbase = Km + (bh * S_ + (size_t)kt * 16) * D_;
    v8f acc = acc_zero();
#pragma unroll
    for (int kc = 0; kc < 2; ++kc) {
      v16bf kf = load_fragB(Kbase + (size_t)r * D_ + kc * 32 + kbB);
      acc = wmma_bf16(qf[kc], kf, acc);
    }
    int key = kt * 16 + nIdx;
    float msk = padmask[(size_t)bb * S_ + key] ? -1e30f : 0.0f;
#pragma unroll
    for (int e = 0; e < 8; ++e)
      sS[(size_t)(mBase + e) * S_ + key] = acc[e] * 0.125f + msk;
  }
  __syncthreads();

  // ---- softmax: 16 threads per row ----
  const int row = tid >> 4;
  const int l16 = tid & 15;
  float lmax = -1e30f;
  for (int c = l16; c < S_; c += 16) lmax = fmaxf(lmax, sS[(size_t)row * S_ + c]);
#pragma unroll
  for (int off = 1; off < 16; off <<= 1) lmax = fmaxf(lmax, __shfl_xor(lmax, off, 16));
  float lsum = 0.0f;
  for (int c = l16; c < S_; c += 16) {
    float e = __expf(sS[(size_t)row * S_ + c] - lmax);
    lsum += e;
    sP[(size_t)row * S_ + c] = (bf16_t)e;  // unnormalized; divide by sum at the end
  }
#pragma unroll
  for (int off = 1; off < 16; off <<= 1) lsum += __shfl_xor(lsum, off, 16);
  if (l16 == 0) sRow[row] = lsum;
  __syncthreads();

  // ---- P x V: each wave accumulates 16x64 over its share of keys ----
  v8f cacc[4];
#pragma unroll
  for (int j = 0; j < 4; ++j) cacc[j] = acc_zero();
  for (int ck = wid; ck < S_ / 32; ck += 8) {
    v16bf pf = load_fragA(&sP[(size_t)r * S_ + ck * 32 + kbA]);
#pragma unroll
    for (int j = 0; j < 4; ++j) {
      const bf16_t* Vbase = Vt + (bh * D_ + j * 16) * S_;
      v16bf vf = load_fragB(Vbase + (size_t)r * S_ + ck * 32 + kbB);
      cacc[j] = wmma_bf16(pf, vf, cacc[j]);
    }
  }
  __syncthreads();  // sS no longer needed as scores -> reuse as staging
#pragma unroll
  for (int j = 0; j < 4; ++j)
#pragma unroll
    for (int e = 0; e < 8; ++e)
      sS[(size_t)wid * (16 * D_) + (mBase + e) * D_ + j * 16 + nIdx] = cacc[j][e];
  __syncthreads();

  // ---- deterministic cross-wave reduce, normalize, store [B,S,C] bf16 ----
  for (int i = tid; i < 16 * D_; i += 256) {
    int m = i >> 6, d = i & 63;
    float v = 0.0f;
#pragma unroll
    for (int w = 0; w < 8; ++w) v += sS[(size_t)w * (16 * D_) + i];
    v /= sRow[m];
    ctx[((size_t)bb * S_ + q0 + m) * C_ + hh * D_ + d] = (bf16_t)v;
  }
}

// ---------------------------------------------------------------------------
// Conv1d-FFN as 9-tap GEMM: h3[g,f] = gelu((sum_t sum_c X[g+t-4,c]*W[t][f][c]
//                                           + bias[f]) / 3)
// Block tile 128(tokens) x 128(F), zero padding at sequence edges.
// ---------------------------------------------------------------------------
__global__ __launch_bounds__(256) void conv_gemm_kernel(
    const bf16_t* __restrict__ Xin,   // [B*S, C]
    const bf16_t* __restrict__ Wk,    // [K][F][C]
    const float* __restrict__ bias,   // [F]
    bf16_t* __restrict__ Hout) {      // [B*S, F]
  __shared__ bf16_t sA[128][40];
  __shared__ bf16_t sB[128][40];
  const int tid = threadIdx.x;
  const int wid = tid >> 5, lane = tid & 31;
  const int wm = wid >> 1, wn = wid & 1;
  const int m0 = blockIdx.y * 128, f0 = blockIdx.x * 128;
  const int bIdx = m0 / S_;
  const int s0 = m0 % S_;
  const int r = lane & 15;
  const int kbA = (lane >> 4) * 8, kbB = (lane >> 4) * 8;

  v8f acc[2][4];
#pragma unroll
  for (int i = 0; i < 2; ++i)
#pragma unroll
    for (int j = 0; j < 4; ++j) acc[i][j] = acc_zero();

  const int pr = tid >> 1, pc = (tid & 1) * 16;
  for (int t = 0; t < KW_; ++t) {
    const bf16_t* Wt = Wk + (size_t)t * F4_ * C_;
    for (int c0 = 0; c0 < C_; c0 += 32) {
#pragma unroll
      for (int i = 0; i < 2; ++i) {
        int cid = tid + 256 * i;
        int rr = cid >> 2, cc = (cid & 3) << 3;
        int sp = s0 + rr + t - 4;
        float4 val;
        if (sp >= 0 && sp < S_)
          val = *reinterpret_cast<const float4*>(
              Xin + ((size_t)bIdx * S_ + sp) * C_ + c0 + cc);
        else
          val = make_float4(0.0f, 0.0f, 0.0f, 0.0f);
        *reinterpret_cast<float4*>(&sA[rr][cc]) = val;
        *reinterpret_cast<float4*>(&sB[rr][cc]) =
            *reinterpret_cast<const float4*>(Wt + (size_t)(f0 + rr) * C_ + c0 + cc);
      }
      if (c0 + 32 < C_)
        __builtin_prefetch(Wt + (size_t)(f0 + pr) * C_ + c0 + 32 + pc, 0, 1);
      __syncthreads();
      v16bf af[2], bfm[4];
#pragma unroll
      for (int i = 0; i < 2; ++i) af[i] = load_fragA(&sA[wm * 32 + i * 16 + r][kbA]);
#pragma unroll
      for (int j = 0; j < 4; ++j) bfm[j] = load_fragB(&sB[wn * 64 + j * 16 + r][kbB]);
#pragma unroll
      for (int i = 0; i < 2; ++i)
#pragma unroll
        for (int j = 0; j < 4; ++j) acc[i][j] = wmma_bf16(af[i], bfm[j], acc[i][j]);
      __syncthreads();
    }
  }

  const int nIdx = lane & 15, mBase = (lane >> 4) * 8;
#pragma unroll
  for (int i = 0; i < 2; ++i)
#pragma unroll
    for (int j = 0; j < 4; ++j) {
      int ff = f0 + wn * 64 + j * 16 + nIdx;
#pragma unroll
      for (int e = 0; e < 8; ++e) {
        int mm = m0 + wm * 32 + i * 16 + mBase + e;
        float v = (acc[i][j][e] + bias[ff]) * (1.0f / 3.0f);  // * K^-0.5
        float g = 0.5f * v * (1.0f + erff(v * 0.70710678118654752f));
        Hout[(size_t)mm * F4_ + ff] = (bf16_t)g;
      }
    }
}

// ---------------------------------------------------------------------------
// Host-side orchestration
// ---------------------------------------------------------------------------
extern "C" void kernel_launch(void* const* d_in, const int* in_sizes, int n_in,
                              void* d_out, int out_size, void* d_ws, size_t ws_size,
                              hipStream_t stream) {
  (void)in_sizes; (void)n_in; (void)out_size; (void)ws_size;
  const float* x     = (const float*)d_in[0];
  const int*   pmask = (const int*)  d_in[1];
  const float* ln1w  = (const float*)d_in[2];
  const float* ln1b  = (const float*)d_in[3];
  const float* winp  = (const float*)d_in[4];
  const float* woutp = (const float*)d_in[5];
  const float* ln2w  = (const float*)d_in[6];
  const float* ln2b  = (const float*)d_in[7];
  const float* wff1  = (const float*)d_in[8];
  const float* bff1  = (const float*)d_in[9];
  const float* wff2  = (const float*)d_in[10];
  const float* bff2  = (const float*)d_in[11];
  float* out = (float*)d_out;

  uintptr_t p = (uintptr_t)d_ws;
  auto alloc = [&](size_t bytes) -> void* {
    uintptr_t q = (p + 255) & ~(uintptr_t)255;
    p = q + bytes;
    return (void*)q;
  };
  bf16_t* win_bf  = (bf16_t*)alloc((size_t)3 * C_ * C_ * sizeof(bf16_t));
  bf16_t* wout_bf = (bf16_t*)alloc((size_t)C_ * C_ * sizeof(bf16_t));
  bf16_t* w1_bf   = (bf16_t*)alloc((size_t)KW_ * F4_ * C_ * sizeof(bf16_t));
  bf16_t* w2_bf   = (bf16_t*)alloc((size_t)C_ * F4_ * sizeof(bf16_t));
  bf16_t* h_bf    = (bf16_t*)alloc((size_t)BS_ * C_ * sizeof(bf16_t));
  float*  qkv_f   = (float*) alloc((size_t)BS_ * 3 * C_ * sizeof(float));
  bf16_t* Qb      = (bf16_t*)alloc((size_t)BS_ * C_ * sizeof(bf16_t));
  bf16_t* Kb      = (bf16_t*)alloc((size_t)BS_ * C_ * sizeof(bf16_t));
  bf16_t* Vtb     = (bf16_t*)alloc((size_t)BS_ * C_ * sizeof(bf16_t));
  bf16_t* ctx_bf  = (bf16_t*)alloc((size_t)BS_ * C_ * sizeof(bf16_t));
  float*  x1_f    = (float*) alloc((size_t)BS_ * C_ * sizeof(float));
  bf16_t* h2_bf   = (bf16_t*)alloc((size_t)BS_ * C_ * sizeof(bf16_t));
  bf16_t* h3_bf   = (bf16_t*)alloc((size_t)BS_ * F4_ * sizeof(bf16_t));

  // 1. weight conversion / repack
  {
    size_t n = (size_t)3 * C_ * C_;
    f32_to_bf16_kernel<<<(unsigned)((n + 255) / 256), 256, 0, stream>>>(winp, win_bf, n);
    n = (size_t)C_ * C_;
    f32_to_bf16_kernel<<<(unsigned)((n + 255) / 256), 256, 0, stream>>>(woutp, wout_bf, n);
    n = (size_t)C_ * F4_;
    f32_to_bf16_kernel<<<(unsigned)((n + 255) / 256), 256, 0, stream>>>(wff2, w2_bf, n);
    n = (size_t)F4_ * C_ * KW_;
    repack_ffn1_kernel<<<(unsigned)((n + 255) / 256), 256, 0, stream>>>(wff1, w1_bf);
  }
  // 2. LN1
  layernorm_kernel<<<BS_, 256, 0, stream>>>(x, ln1w, ln1b, h_bf);
  // 3. QKV projection: [8192,1024] x [3072,1024]^T
  gemm_bf16_kernel<0><<<dim3(3 * C_ / 128, BS_ / 128), 256, 0, stream>>>(
      h_bf, win_bf, qkv_f, nullptr, nullptr, nullptr, BS_, 3 * C_, C_);
  // 4. RoPE + head split (+ V transpose)
  {
    size_t n = (size_t)B_ * S_ * H_ * (D_ / 2);
    rope_split_kernel<<<(unsigned)((n + 255) / 256), 256, 0, stream>>>(qkv_f, Qb, Kb, Vtb);
  }
  // 5. attention (one WG per (b,h,16-query tile); 192KB LDS)
  attn_kernel<<<dim3(S_ / 16, H_, B_), 256, 0, stream>>>(Qb, Kb, Vtb, pmask, ctx_bf);
  // 6. out-proj + residual + nonpad mask
  gemm_bf16_kernel<1><<<dim3(C_ / 128, BS_ / 128), 256, 0, stream>>>(
      ctx_bf, wout_bf, x1_f, x, nullptr, pmask, BS_, C_, C_);
  // 7. LN2
  layernorm_kernel<<<BS_, 256, 0, stream>>>(x1_f, ln2w, ln2b, h2_bf);
  // 8. conv1d FFN (9-tap GEMM) + bias + scale + GELU
  conv_gemm_kernel<<<dim3(F4_ / 128, BS_ / 128), 256, 0, stream>>>(
      h2_bf, w1_bf, bff1, h3_bf);
  // 9. FFN2 + bias + residual + nonpad mask -> final output
  gemm_bf16_kernel<2><<<dim3(C_ / 128, BS_ / 128), 256, 0, stream>>>(
      h3_bf, w2_bf, out, x1_f, bff2, pmask, BS_, C_, F4_);
}
